// GroupedQueryAttention_33191507263683
// MI455X (gfx1250) — matrix operand order repaired and checked by reading, not compiled, not run on hardware
//
#include <hip/hip_runtime.h>
#include <hip/hip_bf16.h>

typedef __attribute__((ext_vector_type(16))) __bf16 bf16x16;
typedef __attribute__((ext_vector_type(8)))  __bf16 bf16x8;
typedef __attribute__((ext_vector_type(4)))  __bf16 bf16x4;
typedef __attribute__((ext_vector_type(8)))  float  f32x8;
typedef __attribute__((ext_vector_type(4)))  unsigned int u32x4;
typedef __attribute__((ext_vector_type(8)))  int  i32x8;
typedef __attribute__((ext_vector_type(4)))  int  i32x4;

#define NUM_HEADS 32
#define NUM_GROUPS 8
#define HEAD_DIM 64
#define SEQ 2048
#define DMODEL 2048
#define KV_WIDTH (NUM_GROUPS * HEAD_DIM)   // 512
#define Q_WIDTH  (NUM_HEADS * HEAD_DIM)    // 2048

#define LDS_PAD 8
#define LDS_STRIDE (64 + LDS_PAD)          // 144B/row = 128B data + 16B pad

#if __has_builtin(__builtin_amdgcn_tensor_load_to_lds) || \
    __has_builtin(__builtin_amdgcn_tensor_load_to_lds_d2)
#define HAVE_TDM 1
#else
#define HAVE_TDM 0
#endif

static __device__ __forceinline__ f32x8 wmma_bf16(bf16x16 a, bf16x16 b, f32x8 c) {
  return __builtin_amdgcn_wmma_f32_16x16x32_bf16(false, a, false, b, (short)0, c,
                                                 false, false);
}
static __device__ __forceinline__ bf16x16 frag_cat(bf16x8 lo, bf16x8 hi) {
  return __builtin_shufflevector(lo, hi, 0, 1, 2, 3, 4, 5, 6, 7,
                                         8, 9, 10, 11, 12, 13, 14, 15);
}
static __device__ __forceinline__ bf16x16 load_frag_contig(const __bf16* p) {
  return frag_cat(*(const bf16x8*)p, *(const bf16x8*)(p + 8));
}
static __device__ __forceinline__ bf16x16 load_frag_split(const __bf16* p) {
  return frag_cat(*(const bf16x8*)p, *(const bf16x8*)(p + 16));
}
static __device__ __forceinline__ bf16x4 cvt4(float4 v) {
  bf16x4 r;
  r[0] = (__bf16)v.x; r[1] = (__bf16)v.y; r[2] = (__bf16)v.z; r[3] = (__bf16)v.w;
  return r;
}

// ---------------------------------------------------------------------------
// TDM: DMA a 64x64 bf16 tile (row pitch `stride0` elements) into LDS with
// 16B row padding (-> LDS_STRIDE rows). Issue from ONE wave per block.
// Descriptor packing per CDNA5 ISA 8.3/8.4.
// ---------------------------------------------------------------------------
static __device__ __forceinline__ void tdm_load_tile_64x64(
    unsigned lds_byte_off, const __bf16* gsrc,
    unsigned rem_d0, unsigned rem_d1, unsigned long long stride0)
{
#if HAVE_TDM
  unsigned long long ga = (unsigned long long)(size_t)gsrc;
  u32x4 g0;
  g0[0] = 1u;                                        // count=1 (user, valid)
  g0[1] = lds_byte_off;                              // lds_addr
  g0[2] = (unsigned)(ga & 0xFFFFFFFFull);            // global_addr[31:0]
  g0[3] = (unsigned)((ga >> 32) & 0x1FFFFFFull)      // global_addr[56:32]
        | (2u << 30);                                // type = 2 ("image")
  i32x8 g1;
  g1[0] = (int)((1u << 16)        // data_size = 2 bytes
              | (1u << 20)        // pad_enable
              | (4u << 22)        // pad_interval: 32 DWORDs (128B)
              | (3u << 25));      // pad_amount: 4 DWORDs (16B)
  g1[1] = (int)((rem_d0 & 0xFFFFu) << 16);                       // tensor_dim0 lo
  g1[2] = (int)((rem_d0 >> 16) | ((rem_d1 & 0xFFFFu) << 16));    // d0 hi | d1 lo
  g1[3] = (int)((rem_d1 >> 16) | (64u << 16));                   // d1 hi | tile_dim0
  g1[4] = (int)(64u);                                            // tile_dim1 (dim2=0)
  g1[5] = (int)(stride0 & 0xFFFFFFFFull);
  g1[6] = (int)((stride0 >> 32) & 0xFFFFull);                    // stride1 = 0
  g1[7] = 0;
#if __has_builtin(__builtin_amdgcn_tensor_load_to_lds)
  i32x4 z4; z4[0] = 0; z4[1] = 0; z4[2] = 0; z4[3] = 0;
  i32x8 z8;
  for (int q = 0; q < 8; ++q) z8[q] = 0;
  // 6-arg form on this toolchain: (g0, g1, g2, g3, g4, cpol)
  __builtin_amdgcn_tensor_load_to_lds(g0, g1, z4, z4, z8, 0);
#else
  __builtin_amdgcn_tensor_load_to_lds_d2(g0, g1, 0);
#endif
#else
  (void)lds_byte_off; (void)gsrc; (void)rem_d0; (void)rem_d1; (void)stride0;
#endif
}

static __device__ __forceinline__ unsigned lds_off(const void* p) {
  return (unsigned)(size_t)p;   // generic LDS pointer: low 32 bits = LDS offset
}

#if !HAVE_TDM
static __device__ __forceinline__ void copy_tile_bf16(
    __bf16* dst, const __bf16* src, size_t stride, int tid, int nthr) {
  for (int i = tid; i < 64 * 8; i += nthr) {
    int r = i >> 3, c8 = (i & 7) << 3;
    *(bf16x8*)(dst + (size_t)r * LDS_STRIDE + c8) =
        *(const bf16x8*)(src + (size_t)r * stride + c8);
  }
}
#endif

// ---------------------------------------------------------------------------
// fp32 -> bf16 flat convert (n multiple of 4)
// ---------------------------------------------------------------------------
__global__ __launch_bounds__(256) void convert_bf16_kernel(
    const float* __restrict__ in, __bf16* __restrict__ out, int n) {
  int i = (blockIdx.x * 256 + threadIdx.x) * 4;
  if (i < n) *(bf16x4*)(out + i) = cvt4(*(const float4*)(in + i));
}

// ---------------------------------------------------------------------------
// Wt[c][r] = (bf16) W[r][c].  W: [R][C] fp32.  LDS-tiled 64x64.
// ---------------------------------------------------------------------------
__global__ __launch_bounds__(256) void transpose_convert_kernel(
    const float* __restrict__ W, __bf16* __restrict__ Wt, int R, int C) {
  __shared__ __align__(16) __bf16 t64[64][68];
  const int r0 = blockIdx.y * 64, c0 = blockIdx.x * 64;
  const int tid = threadIdx.x;
  for (int i = tid; i < 64 * 16; i += 256) {
    int rr = i >> 4, c4 = (i & 15) << 2;
    float4 v = *(const float4*)(W + (size_t)(r0 + rr) * C + c0 + c4);
    t64[c4 + 0][rr] = (__bf16)v.x;
    t64[c4 + 1][rr] = (__bf16)v.y;
    t64[c4 + 2][rr] = (__bf16)v.z;
    t64[c4 + 3][rr] = (__bf16)v.w;
  }
  __syncthreads();
  for (int i = tid; i < 64 * 16; i += 256) {
    int cc = i >> 4, k4 = (i & 15) << 2;
    bf16x4 pk;
    pk[0] = t64[cc][k4 + 0]; pk[1] = t64[cc][k4 + 1];
    pk[2] = t64[cc][k4 + 2]; pk[3] = t64[cc][k4 + 3];
    *(bf16x4*)(Wt + (size_t)(c0 + cc) * R + r0 + k4) = pk;
  }
}

// ---------------------------------------------------------------------------
// C[M,N] = A[M,K] * B[K,N] + bias[N], with B given TRANSPOSED (Bt[N][K]).
// A,Bt bf16 global; tiles DMA'd by TDM (double-buffered); fp32 accumulate.
// MODE: 0 = bf16 row-major out, 1 = bf16 transposed out, 2 = fp32 out.
// 128 threads = 4 waves; tile 64x64x64.
// ---------------------------------------------------------------------------
template <int MODE>
__global__ __launch_bounds__(128) void gemm_bias_kernel(
    const __bf16* __restrict__ A, const __bf16* __restrict__ Bt,
    const float* __restrict__ bias, void* __restrict__ Cout,
    int M, int N, int K)
{
  __shared__ __align__(16) __bf16 As[2][64][LDS_STRIDE];  // [m][k]
  __shared__ __align__(16) __bf16 Bs[2][64][LDS_STRIDE];  // [n][k]
  const int n0 = blockIdx.x * 64;
  const int m0 = blockIdx.y * 64;
  const int tid  = threadIdx.x;
  const int wave = tid >> 5;
  const int lane = tid & 31;
  const int lh   = lane & 15;
  const int hi   = lane >> 4;

  f32x8 acc[4];
  for (int t = 0; t < 4; ++t)
    for (int r = 0; r < 8; ++r) acc[t][r] = 0.0f;

  const int ntiles = K / 64;
  const int arow = wave * 16 + lh;

#if HAVE_TDM
  if (wave == 0) {
    tdm_load_tile_64x64(lds_off(&As[0][0][0]), A + (size_t)m0 * K,
                        (unsigned)K, (unsigned)(M - m0), (unsigned long long)K);
    tdm_load_tile_64x64(lds_off(&Bs[0][0][0]), Bt + (size_t)n0 * K,
                        (unsigned)K, (unsigned)(N - n0), (unsigned long long)K);
  }
  for (int it = 0; it < ntiles; ++it) {
    const int cur = it & 1;
    if (wave == 0) __builtin_amdgcn_s_wait_tensorcnt(0);
    __syncthreads();           // tile `it` ready; tile `it-1` reads finished
    if (wave == 0 && it + 1 < ntiles) {
      const int k1 = (it + 1) * 64;
      tdm_load_tile_64x64(lds_off(&As[cur ^ 1][0][0]),
                          A + (size_t)m0 * K + k1,
                          (unsigned)(K - k1), (unsigned)(M - m0),
                          (unsigned long long)K);
      tdm_load_tile_64x64(lds_off(&Bs[cur ^ 1][0][0]),
                          Bt + (size_t)n0 * K + k1,
                          (unsigned)(K - k1), (unsigned)(N - n0),
                          (unsigned long long)K);
    }
    for (int kc = 0; kc < 64; kc += 32) {
      bf16x16 af = load_frag_split(&As[cur][arow][kc + hi * 8]);
      for (int t = 0; t < 4; ++t) {
        bf16x16 bfr = load_frag_split(&Bs[cur][t * 16 + lh][kc + hi * 8]);
        acc[t] = wmma_bf16(af, bfr, acc[t]);
      }
    }
  }
#else
  for (int it = 0; it < ntiles; ++it) {
    const int k0 = it * 64;
    __syncthreads();
    copy_tile_bf16(&As[0][0][0], A + (size_t)m0 * K + k0, K, tid, 128);
    copy_tile_bf16(&Bs[0][0][0], Bt + (size_t)n0 * K + k0, K, tid, 128);
    __syncthreads();
    for (int kc = 0; kc < 64; kc += 32) {
      bf16x16 af = load_frag_split(&As[0][arow][kc + hi * 8]);
      for (int t = 0; t < 4; ++t) {
        bf16x16 bfr = load_frag_split(&Bs[0][t * 16 + lh][kc + hi * 8]);
        acc[t] = wmma_bf16(af, bfr, acc[t]);
      }
    }
  }
#endif

  for (int t = 0; t < 4; ++t) {
    int cg = n0 + t * 16 + lh;
    float bv = bias[cg];
    for (int r = 0; r < 8; ++r) {
      int rg = m0 + wave * 16 + hi * 8 + r;
      float v = acc[t][r] + bv;
      if (MODE == 0) ((__bf16*)Cout)[(size_t)rg * N + cg] = (__bf16)v;
      else if (MODE == 1) ((__bf16*)Cout)[(size_t)cg * M + rg] = (__bf16)v;
      else ((float*)Cout)[(size_t)rg * N + cg] = v;
    }
  }
}

// ---------------------------------------------------------------------------
// Flash attention (causal, GQA 4:1). Grid: (SEQ/64, NUM_HEADS); 4 waves.
// Inputs bf16: Q [S][2048], K [S][512], V TRANSPOSED [512][S].
// K/V tiles DMA'd by TDM (double-buffered). Output Ab bf16 [S][2048].
// ---------------------------------------------------------------------------
__global__ __launch_bounds__(128) void flash_attn_kernel(
    const __bf16* __restrict__ Q, const __bf16* __restrict__ Kp,
    const __bf16* __restrict__ Vt, __bf16* __restrict__ O)
{
  __shared__ __align__(16) __bf16 Ks[2][64][LDS_STRIDE];   // [key][d]
  __shared__ __align__(16) __bf16 Vs[2][64][LDS_STRIDE];   // [dv][key]
  __shared__ __align__(16) __bf16 Ps[4][16][LDS_STRIDE];   // per-wave [row][key]

  const int qt   = blockIdx.x;
  const int head = blockIdx.y;
  const int grp  = head / (NUM_HEADS / NUM_GROUPS);
  const int tid  = threadIdx.x;
  const int wave = tid >> 5;
  const int lane = tid & 31;
  const int lh   = lane & 15;
  const int hi   = lane >> 4;
  const float scale = 0.125f;

  // Q fragments (A-layout), bf16 directly from global.
  bf16x16 qf[2];
  {
    const __bf16* qrow =
        Q + (size_t)(qt * 64 + wave * 16 + lh) * Q_WIDTH + head * HEAD_DIM;
    for (int c = 0; c < 2; ++c) {
      int kb = c * 32 + hi * 8;
      qf[c] = frag_cat(*(const bf16x8*)(qrow + kb),
                       *(const bf16x8*)(qrow + kb + 16));
    }
  }

  f32x8 o[4];
  for (int t = 0; t < 4; ++t)
    for (int r = 0; r < 8; ++r) o[t][r] = 0.0f;
  float mr[8], lr[8];
  for (int r = 0; r < 8; ++r) { mr[r] = -1e30f; lr[r] = 0.0f; }

#if HAVE_TDM
  if (wave == 0) {
    tdm_load_tile_64x64(lds_off(&Ks[0][0][0]),
                        Kp + (size_t)0 * KV_WIDTH + grp * HEAD_DIM,
                        (unsigned)(KV_WIDTH - grp * HEAD_DIM), (unsigned)SEQ,
                        (unsigned long long)KV_WIDTH);
    tdm_load_tile_64x64(lds_off(&Vs[0][0][0]),
                        Vt + (size_t)(grp * HEAD_DIM) * SEQ,
                        (unsigned)SEQ, (unsigned)(KV_WIDTH - grp * HEAD_DIM),
                        (unsigned long long)SEQ);
  }
#endif

  for (int kt = 0; kt <= qt; ++kt) {
    const int cur = kt & 1;
#if HAVE_TDM
    if (wave == 0) __builtin_amdgcn_s_wait_tensorcnt(0);
    __syncthreads();
    if (wave == 0 && kt + 1 <= qt) {
      const int kk = (kt + 1) * 64;
      tdm_load_tile_64x64(lds_off(&Ks[cur ^ 1][0][0]),
                          Kp + (size_t)kk * KV_WIDTH + grp * HEAD_DIM,
                          (unsigned)(KV_WIDTH - grp * HEAD_DIM),
                          (unsigned)(SEQ - kk), (unsigned long long)KV_WIDTH);
      tdm_load_tile_64x64(lds_off(&Vs[cur ^ 1][0][0]),
                          Vt + (size_t)(grp * HEAD_DIM) * SEQ + kk,
                          (unsigned)(SEQ - kk),
                          (unsigned)(KV_WIDTH - grp * HEAD_DIM),
                          (unsigned long long)SEQ);
    }
#else
    __syncthreads();
    copy_tile_bf16(&Ks[0][0][0], Kp + (size_t)(kt * 64) * KV_WIDTH + grp * HEAD_DIM,
                   KV_WIDTH, tid, 128);
    copy_tile_bf16(&Vs[0][0][0], Vt + (size_t)(grp * HEAD_DIM) * SEQ + kt * 64,
                   SEQ, tid, 128);
    __syncthreads();
#endif

    // S = Q K^T (B[d][key] = Ks[key][d] -> split row fragment)
    f32x8 s[4];
    for (int t = 0; t < 4; ++t)
      for (int r = 0; r < 8; ++r) s[t][r] = 0.0f;
    for (int c = 0; c < 2; ++c) {
      int kc = c * 32;
      for (int t = 0; t < 4; ++t) {
        bf16x16 bfr = load_frag_split(
            &Ks[HAVE_TDM ? cur : 0][t * 16 + lh][kc + hi * 8]);
        s[t] = wmma_bf16(qf[c], bfr, s[t]);
      }
    }

    const bool diag = (kt == qt);
    for (int t = 0; t < 4; ++t) {
      int col = t * 16 + lh;
      for (int r = 0; r < 8; ++r) {
        float v = s[t][r] * scale;
        if (diag && col > (wave * 16 + hi * 8 + r)) v = -1e30f;
        s[t][r] = v;
      }
    }

    // Online softmax (row stats per register index, replicated per half)
    for (int r = 0; r < 8; ++r) {
      float x = fmaxf(fmaxf(s[0][r], s[1][r]), fmaxf(s[2][r], s[3][r]));
      for (int off = 1; off < 16; off <<= 1) x = fmaxf(x, __shfl_xor(x, off, 32));
      float mn = fmaxf(mr[r], x);
      float sum = 0.0f;
      for (int t = 0; t < 4; ++t) {
        float p = __expf(s[t][r] - mn);
        s[t][r] = p;
        sum += p;
      }
      for (int off = 1; off < 16; off <<= 1) sum += __shfl_xor(sum, off, 32);
      float f = __expf(mr[r] - mn);
      lr[r] = lr[r] * f + sum;
      mr[r] = mn;
      for (int t = 0; t < 4; ++t) o[t][r] *= f;
      for (int t = 0; t < 4; ++t)
        Ps[wave][hi * 8 + r][t * 16 + lh] = (__bf16)s[t][r];
    }
    __syncthreads();

    // O += P V
    for (int c = 0; c < 2; ++c) {
      int kc = c * 32;
      bf16x16 af = load_frag_split(&Ps[wave][lh][kc + hi * 8]);
      for (int t = 0; t < 4; ++t) {
        bf16x16 bfr = load_frag_split(
            &Vs[HAVE_TDM ? cur : 0][t * 16 + lh][kc + hi * 8]);
        o[t] = wmma_bf16(af, bfr, o[t]);
      }
    }
  }

  for (int t = 0; t < 4; ++t) {
    int cg = head * HEAD_DIM + t * 16 + lh;
    for (int r = 0; r < 8; ++r) {
      int rg = qt * 64 + wave * 16 + hi * 8 + r;
      O[(size_t)rg * Q_WIDTH + cg] = (__bf16)(o[t][r] / lr[r]);
    }
  }
}

// ---------------------------------------------------------------------------
extern "C" void kernel_launch(void* const* d_in, const int* in_sizes, int n_in,
                              void* d_out, int out_size, void* d_ws, size_t ws_size,
                              hipStream_t stream) {
  (void)in_sizes; (void)n_in; (void)out_size; (void)ws_size;
  const float* queries = (const float*)d_in[0];
  const float* keys    = (const float*)d_in[1];
  const float* values  = (const float*)d_in[2];
  const float* Wq      = (const float*)d_in[3];
  const float* bq      = (const float*)d_in[4];
  const float* Wk      = (const float*)d_in[5];
  const float* bk      = (const float*)d_in[6];
  const float* Wv      = (const float*)d_in[7];
  const float* bv      = (const float*)d_in[8];
  const float* Wo      = (const float*)d_in[9];
  const float* bo      = (const float*)d_in[10];
  float* out = (float*)d_out;

  char* ws = (char*)d_ws;
  size_t off = 0;
  auto carve = [&](size_t elems) {
    __bf16* p = (__bf16*)(ws + off);
    off += ((elems * 2) + 255) & ~(size_t)255;
    return p;
  };
  __bf16* qbf = carve((size_t)SEQ * DMODEL);
  __bf16* kbf = carve((size_t)SEQ * DMODEL);
  __bf16* vbf = carve((size_t)SEQ * DMODEL);
  __bf16* WqT = carve((size_t)DMODEL * Q_WIDTH);
  __bf16* WkT = carve((size_t)DMODEL * KV_WIDTH);
  __bf16* WvT = carve((size_t)DMODEL * KV_WIDTH);
  __bf16* WoT = carve((size_t)Q_WIDTH * DMODEL);
  __bf16* Qb  = carve((size_t)SEQ * Q_WIDTH);
  __bf16* Kb  = carve((size_t)SEQ * KV_WIDTH);
  __bf16* Vtb = carve((size_t)KV_WIDTH * SEQ);
  __bf16* Ab  = carve((size_t)SEQ * Q_WIDTH);

  const int nAct = SEQ * DMODEL;
  convert_bf16_kernel<<<nAct / 4 / 256, 256, 0, stream>>>(queries, qbf, nAct);
  convert_bf16_kernel<<<nAct / 4 / 256, 256, 0, stream>>>(keys, kbf, nAct);
  convert_bf16_kernel<<<nAct / 4 / 256, 256, 0, stream>>>(values, vbf, nAct);

  transpose_convert_kernel<<<dim3(Q_WIDTH / 64, DMODEL / 64), 256, 0, stream>>>(
      Wq, WqT, DMODEL, Q_WIDTH);
  transpose_convert_kernel<<<dim3(KV_WIDTH / 64, DMODEL / 64), 256, 0, stream>>>(
      Wk, WkT, DMODEL, KV_WIDTH);
  transpose_convert_kernel<<<dim3(KV_WIDTH / 64, DMODEL / 64), 256, 0, stream>>>(
      Wv, WvT, DMODEL, KV_WIDTH);
  transpose_convert_kernel<<<dim3(DMODEL / 64, Q_WIDTH / 64), 256, 0, stream>>>(
      Wo, WoT, Q_WIDTH, DMODEL);

  gemm_bias_kernel<0><<<dim3(Q_WIDTH / 64, SEQ / 64), 128, 0, stream>>>(
      qbf, WqT, bq, Qb, SEQ, Q_WIDTH, DMODEL);
  gemm_bias_kernel<0><<<dim3(KV_WIDTH / 64, SEQ / 64), 128, 0, stream>>>(
      kbf, WkT, bk, Kb, SEQ, KV_WIDTH, DMODEL);
  gemm_bias_kernel<1><<<dim3(KV_WIDTH / 64, SEQ / 64), 128, 0, stream>>>(
      vbf, WvT, bv, Vtb, SEQ, KV_WIDTH, DMODEL);

  flash_attn_kernel<<<dim3(SEQ / 64, NUM_HEADS), 128, 0, stream>>>(Qb, Kb, Vtb, Ab);

  gemm_bias_kernel<2><<<dim3(DMODEL / 64, SEQ / 64), 128, 0, stream>>>(
      Ab, WoT, bo, out, SEQ, DMODEL, Q_WIDTH);
}
